// TopKSAE_46840913330330
// MI455X (gfx1250) — matrix operand order repaired and checked by reading, not compiled, not run on hardware
//
#include <hip/hip_runtime.h>

// ---------------- problem constants ----------------
#define B_ROWS   16384
#define D_IN     768
#define D_SAE    12288
#define TOPK     40

// WMMA vector types (CDNA5 gfx1250, wave32)
typedef __attribute__((ext_vector_type(16))) __bf16  v16bf;
typedef __attribute__((ext_vector_type(8)))  float   v8f;

// LDS row stride: 40 bf16 = 80 bytes (16B-aligned; 20-bank stride -> 16 lanes conflict-free)
#define LDS_STRIDE   40
#define REGION_ELTS  (128 * LDS_STRIDE)   // one 128x32 bf16 tile (padded)
#define REGION_BYTES (REGION_ELTS * 2)    // 10240 B

#ifndef __has_builtin
#define __has_builtin(x) 0
#endif
#if __has_builtin(__builtin_amdgcn_tensor_load_to_lds) && __has_builtin(__builtin_amdgcn_s_wait_tensorcnt)
#define HAVE_TDM 1
#else
#define HAVE_TDM 0
#endif

// ---------------- helpers ----------------
__device__ __forceinline__ unsigned short f2bf(float f) {        // RNE f32 -> bf16
  unsigned u = __float_as_uint(f);
  unsigned r = u + 0x7FFFu + ((u >> 16) & 1u);
  return (unsigned short)(r >> 16);
}
__device__ __forceinline__ float bf2f(unsigned short s) {
  return __uint_as_float(((unsigned)s) << 16);
}
__device__ __forceinline__ unsigned f2key(float f) {             // order-preserving key
  unsigned u = __float_as_uint(f);
  return (u & 0x80000000u) ? ~u : (u | 0x80000000u);
}
__device__ __forceinline__ float key2f(unsigned k) {
  unsigned u = (k & 0x80000000u) ? (k ^ 0x80000000u) : ~k;
  return __uint_as_float(u);
}

#if HAVE_TDM
typedef __attribute__((ext_vector_type(4))) unsigned int u32x4;
typedef __attribute__((ext_vector_type(4))) int          i32x4;
typedef __attribute__((ext_vector_type(8))) int          i32x8;

// 2D TDM tile load: 128 rows x 64 B (32 bf16), row stride 768 elements, with
// hardware LDS padding: 16 B pad after every 16 dwords -> 80 B LDS row stride.
__device__ __forceinline__ void tdm_load_2d(unsigned lds_off, const void* gptr, unsigned rows) {
  const unsigned long long ga = (unsigned long long)(size_t)gptr;
  u32x4 g0;
  g0[0] = 1u;                                              // count=1 (user descriptor)
  g0[1] = lds_off;                                         // lds_addr (bytes)
  g0[2] = (unsigned)ga;                                    // global_addr[31:0]
  g0[3] = (unsigned)((ga >> 32) & 0x01FFFFFFull)           // global_addr[56:32]
        | (2u << 30);                                      // type = 2 ("image")
  i32x8 g1;
  g1[0] = (int)((1u << 16)                                 // data_size = 2 bytes
              | (1u << 20)                                 // pad_enable
              | (3u << 22)                                 // pad_interval: 16 dwords (64 B)
              | (3u << 25));                               // pad_amount:   4 dwords (16 B)
  g1[1] = (int)(768u << 16);                               // tensor_dim0 = 768 (bits 63:48)
  g1[2] = (int)((rows & 0xFFFFu) << 16);                   // dim0 hi=0 | tensor_dim1 lo
  g1[3] = (int)((rows >> 16) | (32u << 16));               // tensor_dim1 hi | tile_dim0 = 32
  g1[4] = 128;                                             // tile_dim1 = 128, tile_dim2 = 0
  g1[5] = 768;                                             // tensor_dim0_stride = 768
  g1[6] = 0;
  g1[7] = 0;
  const i32x4 z4 = {0, 0, 0, 0};                           // groups 2/3 unused (2D tile)
#if defined(__clang_major__) && (__clang_major__ >= 23)
  const i32x8 z8 = {0, 0, 0, 0, 0, 0, 0, 0};
  __builtin_amdgcn_tensor_load_to_lds(g0, g1, z4, z4, z8, 0);
#else
  __builtin_amdgcn_tensor_load_to_lds(g0, g1, z4, z4, 0);
#endif
}
#endif

// ================================================================
// Kernel 0a: x (f32) -> x_hi/x_lo bf16 split, same row-major layout
// ================================================================
__global__ __launch_bounds__(256) void sae_cvt_x(const float* __restrict__ x,
    unsigned short* __restrict__ xh, unsigned short* __restrict__ xl)
{
  const size_t i = (size_t)blockIdx.x * 256 + threadIdx.x;   // float4 index
  const float4 v = ((const float4*)x)[i];
  const float vv[4] = {v.x, v.y, v.z, v.w};
  union { unsigned short s[4]; uint2 u; } ph, pl;
  #pragma unroll
  for (int j = 0; j < 4; ++j) {
    const unsigned short hb = f2bf(vv[j]);
    ph.s[j] = hb;
    pl.s[j] = f2bf(vv[j] - bf2f(hb));
  }
  ((uint2*)xh)[i] = ph.u;
  ((uint2*)xl)[i] = pl.u;
}

// ================================================================
// Kernel 0b: W_enc (D_IN x D_SAE f32) -> transposed Wt_hi/Wt_lo (D_SAE x D_IN bf16)
// 32x32 LDS tile transpose, coalesced both sides
// ================================================================
__global__ __launch_bounds__(256) void sae_cvt_wt(const float* __restrict__ W,
    unsigned short* __restrict__ wh, unsigned short* __restrict__ wl)
{
  __shared__ float tile[32][33];
  const int t  = threadIdx.x;
  const int k0 = blockIdx.x * 32;       // K (768)
  const int n0 = blockIdx.y * 32;       // N (12288)
  {
    const int kk = t >> 3, c4 = (t & 7) * 4;
    const float4 v = *(const float4*)(W + (size_t)(k0 + kk) * D_SAE + n0 + c4);
    tile[kk][c4 + 0] = v.x; tile[kk][c4 + 1] = v.y;
    tile[kk][c4 + 2] = v.z; tile[kk][c4 + 3] = v.w;
  }
  __syncthreads();
  {
    const int n = t >> 3, k4 = (t & 7) * 4;
    union { unsigned short s[4]; uint2 u; } ph, pl;
    #pragma unroll
    for (int j = 0; j < 4; ++j) {
      const float f = tile[k4 + j][n];
      const unsigned short hb = f2bf(f);
      ph.s[j] = hb;
      pl.s[j] = f2bf(f - bf2f(hb));
    }
    *(uint2*)(wh + (size_t)(n0 + n) * D_IN + k0 + k4) = ph.u;
    *(uint2*)(wl + (size_t)(n0 + n) * D_IN + k0 + k4) = pl.u;
  }
}

// ================================================================
// Kernel 1: encode GEMM  h = x @ W_enc + b_enc  (bf16-split WMMA)
// 128x128 tile / 256-thread block; TDM double-buffered LDS staging.
// LDS: 8 dynamic regions of 10240 B = 80 KB ([buf][Ah,Al,Bh,Bl])
// ================================================================
__global__ __launch_bounds__(256) void sae_encode_wmma(
    const unsigned short* __restrict__ xh, const unsigned short* __restrict__ xl,
    const unsigned short* __restrict__ wh, const unsigned short* __restrict__ wl,
    const float* __restrict__ b_enc, float* __restrict__ h)
{
  extern __shared__ unsigned short smem[];   // 8 * REGION_ELTS
  const int tid  = threadIdx.x;
  const int lane = tid & 31;
  const int wid  = tid >> 5;
  const int wm   = wid >> 1;                 // 0..3 -> 32-row sub-block
  const int wn   = wid & 1;                  // 0..1 -> 64-col sub-block
  const int row0 = blockIdx.y * 128;
  const int col0 = blockIdx.x * 128;
  // ISA 10.2: flat LDS-aperture address[31:0] == LDS byte offset
  const unsigned lds0 = (unsigned)(size_t)smem;

  v8f acc[2][4];
  #pragma unroll
  for (int i = 0; i < 2; ++i)
    #pragma unroll
    for (int j = 0; j < 4; ++j)
      acc[i][j] = {};

#if HAVE_TDM
  auto issue = [&](int buf, int kt) {
    const int k0 = kt * 32;
    tdm_load_2d(lds0 + (unsigned)(buf * 4 + 0) * REGION_BYTES, xh + (size_t)row0 * D_IN + k0, B_ROWS);
    tdm_load_2d(lds0 + (unsigned)(buf * 4 + 1) * REGION_BYTES, xl + (size_t)row0 * D_IN + k0, B_ROWS);
    tdm_load_2d(lds0 + (unsigned)(buf * 4 + 2) * REGION_BYTES, wh + (size_t)col0 * D_IN + k0, D_SAE);
    tdm_load_2d(lds0 + (unsigned)(buf * 4 + 3) * REGION_BYTES, wl + (size_t)col0 * D_IN + k0, D_SAE);
  };
  if (wid == 0) issue(0, 0);
#endif

  for (int kt = 0; kt < D_IN / 32; ++kt) {
    const int cur = kt & 1;
#if HAVE_TDM
    if (wid == 0) {
      if (kt + 1 < D_IN / 32) {
        issue(cur ^ 1, kt + 1);                    // prefetch next buffer (async DMA)
        __builtin_amdgcn_s_wait_tensorcnt(4);      // in-order: current buffer complete
      } else {
        __builtin_amdgcn_s_wait_tensorcnt(0);
      }
    }
#else
    { // fallback: bf16 uint4 global loads -> LDS stores (same padded layout)
      const int k0 = kt * 32;
      unsigned short* dAh = smem + (cur * 4 + 0) * REGION_ELTS;
      unsigned short* dAl = smem + (cur * 4 + 1) * REGION_ELTS;
      unsigned short* dBh = smem + (cur * 4 + 2) * REGION_ELTS;
      unsigned short* dBl = smem + (cur * 4 + 3) * REGION_ELTS;
      #pragma unroll
      for (int q = 0; q < 2; ++q) {
        const int id  = tid * 2 + q;      // 0..511
        const int r   = id >> 2;          // row/col index 0..127
        const int seg = (id & 3) * 8;     // 8-element (16B) segment within 64B row
        *(uint4*)(dAh + r * LDS_STRIDE + seg) = *(const uint4*)(xh + (size_t)(row0 + r) * D_IN + k0 + seg);
        *(uint4*)(dAl + r * LDS_STRIDE + seg) = *(const uint4*)(xl + (size_t)(row0 + r) * D_IN + k0 + seg);
        *(uint4*)(dBh + r * LDS_STRIDE + seg) = *(const uint4*)(wh + (size_t)(col0 + r) * D_IN + k0 + seg);
        *(uint4*)(dBl + r * LDS_STRIDE + seg) = *(const uint4*)(wl + (size_t)(col0 + r) * D_IN + k0 + seg);
      }
    }
#endif
    __syncthreads();   // current buffer visible to all waves

    const unsigned short* sAh = smem + (cur * 4 + 0) * REGION_ELTS;
    const unsigned short* sAl = smem + (cur * 4 + 1) * REGION_ELTS;
    const unsigned short* sBh = smem + (cur * 4 + 2) * REGION_ELTS;
    const unsigned short* sBl = smem + (cur * 4 + 3) * REGION_ELTS;

    // A 16x32 bf16 frag: lane m=L&15; lanes 0-15 K 0-7 & 16-23, lanes 16-31 K 8-15 & 24-31
    v16bf ah[2], al[2], bh[4], bl[4];
    #pragma unroll
    for (int i = 0; i < 2; ++i) {
      const int rbase = (wm * 32 + i * 16 + (lane & 15)) * LDS_STRIDE + ((lane < 16) ? 0 : 8);
      ((uint4*)&ah[i])[0] = *(const uint4*)&sAh[rbase];
      ((uint4*)&ah[i])[1] = *(const uint4*)&sAh[rbase + 16];
      ((uint4*)&al[i])[0] = *(const uint4*)&sAl[rbase];
      ((uint4*)&al[i])[1] = *(const uint4*)&sAl[rbase + 16];
    }
    // B 32x16 bf16 frag: lane n=L&15; lanes 0-15 K 0-7 & 16-23, lanes 16-31 K 8-15 & 24-31
    #pragma unroll
    for (int j = 0; j < 4; ++j) {
      const int cbase = (wn * 64 + j * 16 + (lane & 15)) * LDS_STRIDE + ((lane < 16) ? 0 : 8);
      ((uint4*)&bh[j])[0] = *(const uint4*)&sBh[cbase];
      ((uint4*)&bh[j])[1] = *(const uint4*)&sBh[cbase + 16];
      ((uint4*)&bl[j])[0] = *(const uint4*)&sBl[cbase];
      ((uint4*)&bl[j])[1] = *(const uint4*)&sBl[cbase + 16];
    }

    // 3-term bf16 split: hh + hl + lh  (~fp32-accurate product)
    #pragma unroll
    for (int i = 0; i < 2; ++i)
      #pragma unroll
      for (int j = 0; j < 4; ++j) {
        acc[i][j] = __builtin_amdgcn_wmma_f32_16x16x32_bf16(false, ah[i], false, bh[j],
                                                            (short)0, acc[i][j], false, false);
        acc[i][j] = __builtin_amdgcn_wmma_f32_16x16x32_bf16(false, ah[i], false, bl[j],
                                                            (short)0, acc[i][j], false, false);
        acc[i][j] = __builtin_amdgcn_wmma_f32_16x16x32_bf16(false, al[i], false, bh[j],
                                                            (short)0, acc[i][j], false, false);
      }
    __syncthreads();   // all reads of `cur` done before it is refilled
  }

  // epilogue: + b_enc, store h. C/D layout: VGPR r = row r (lanes 0-15) / 8+r (lanes 16-31)
  #pragma unroll
  for (int i = 0; i < 2; ++i) {
    const int mrow = row0 + wm * 32 + i * 16 + ((lane < 16) ? 0 : 8);
    #pragma unroll
    for (int j = 0; j < 4; ++j) {
      const int coln = col0 + wn * 64 + j * 16 + (lane & 15);
      const float bb = b_enc[coln];
      #pragma unroll
      for (int r = 0; r < 8; ++r)
        h[(size_t)(mrow + r) * D_SAE + coln] = acc[i][j][r] + bb;
    }
  }
}

// ================================================================
// Kernel 2: per-row top-40 (+ReLU) via 4-pass MSB-first radix select
// (per-wave 256-bin LDS histograms -> 8x fewer full-array scans than
// bisection), in-place sparsify, compact (idx,val) list out.
// ================================================================
__global__ __launch_bounds__(256) void sae_topk(
    float* __restrict__ codes, float* __restrict__ nz_val, int* __restrict__ nz_idx)
{
  __shared__ unsigned keys[D_SAE];     // 48 KB
  __shared__ int hist[8][256];         // per-wave histograms, 8 KB
  __shared__ int tie_ctr, out_ctr, sh_r;
  __shared__ unsigned sh_prefix;

  const int tid = threadIdx.x;
  const int w   = tid >> 5;
  const int row = blockIdx.x;
  float* hrow = codes + (size_t)row * D_SAE;

  for (int i = tid; i < D_SAE / 4; i += 256) {
    const float4 v = ((const float4*)hrow)[i];
    keys[i * 4 + 0] = f2key(v.x);
    keys[i * 4 + 1] = f2key(v.y);
    keys[i * 4 + 2] = f2key(v.z);
    keys[i * 4 + 3] = f2key(v.w);
  }
  if (tid == 0) { tie_ctr = 0; out_ctr = 0; }
  __syncthreads();

  // ---- radix select: find K-th largest key, MSB digit first ----
  unsigned prefix = 0u, known = 0u;    // fixed high bits / their mask
  int r = TOPK;                        // rank remaining within current prefix class
  #pragma unroll
  for (int p = 0; p < 4; ++p) {
    const int shift = 24 - p * 8;
    for (int i = tid; i < 8 * 256; i += 256) ((int*)hist)[i] = 0;
    __syncthreads();
    for (int i = tid; i < D_SAE; i += 256) {
      const unsigned kk = keys[i];
      if ((kk & known) == prefix) atomicAdd(&hist[w][(kk >> shift) & 0xFFu], 1);
    }
    __syncthreads();
    for (int d = tid; d < 256; d += 256) {          // fold 8 wave-histograms
      int s = 0;
      #pragma unroll
      for (int ww = 0; ww < 8; ++ww) s += hist[ww][d];
      hist[0][d] = s;
    }
    __syncthreads();
    if (tid == 0) {                                  // suffix scan from top digit
      int cum = 0, rr = r;
      for (int d = 255; d >= 0; --d) {
        const int c = hist[0][d];
        if (cum + c >= rr) { sh_prefix = prefix | ((unsigned)d << shift); sh_r = rr - cum; break; }
        cum += c;
      }
    }
    __syncthreads();
    prefix = sh_prefix;
    r      = sh_r;
    known |= (0xFFu << shift);
    __syncthreads();
  }
  const unsigned T = prefix;           // exact key of K-th largest
  const int ties_needed = r;           // # of keys == T to keep

  // ---- selection + in-place write + compact list ----
  for (int i = tid; i < D_SAE; i += 256) {
    const unsigned kk = keys[i];
    float outv = 0.0f;
    bool sel = (kk > T);
    if (!sel && kk == T) sel = (atomicAdd(&tie_ctr, 1) < ties_needed);
    if (sel) {
      const float f = key2f(kk);
      outv = f > 0.0f ? f : 0.0f;      // ReLU on kept values
      const int slot = atomicAdd(&out_ctr, 1);
      if (slot < TOPK) { nz_idx[row * TOPK + slot] = i; nz_val[row * TOPK + slot] = outv; }
    }
    hrow[i] = outv;
  }
  __syncthreads();
  if (tid == 0)
    for (int s = out_ctr; s < TOPK; ++s) { nz_idx[row * TOPK + s] = 0; nz_val[row * TOPK + s] = 0.0f; }
}

// ================================================================
// Kernel 3: sparse decode  recon = codes @ W_dec + b_dec (W_dec L2-resident)
// ================================================================
__global__ __launch_bounds__(192) void sae_decode(
    const float* __restrict__ W_dec, const float* __restrict__ b_dec,
    const float* __restrict__ nz_val, const int* __restrict__ nz_idx,
    float* __restrict__ recon)
{
  __shared__ int   sidx[TOPK];
  __shared__ float sval[TOPK];
  const int tid = threadIdx.x;   // 192 threads x float4 = 768 cols
  const int row = blockIdx.x;
  if (tid < TOPK) { sidx[tid] = nz_idx[row * TOPK + tid]; sval[tid] = nz_val[row * TOPK + tid]; }
  __syncthreads();

  float4 acc = ((const float4*)b_dec)[tid];
  #pragma unroll 8
  for (int j = 0; j < TOPK; ++j) {
    const float  v = sval[j];
    const float4 w = *(const float4*)(W_dec + (size_t)sidx[j] * D_IN + tid * 4);
    acc.x += v * w.x; acc.y += v * w.y; acc.z += v * w.z; acc.w += v * w.w;
  }
  ((float4*)(recon + (size_t)row * D_IN))[tid] = acc;
}

// ================================================================
extern "C" void kernel_launch(void* const* d_in, const int* in_sizes, int n_in,
                              void* d_out, int out_size, void* d_ws, size_t ws_size,
                              hipStream_t stream) {
  (void)in_sizes; (void)n_in; (void)out_size; (void)ws_size;
  const float* x     = (const float*)d_in[0];
  const float* W_enc = (const float*)d_in[1];
  const float* b_enc = (const float*)d_in[2];
  const float* W_dec = (const float*)d_in[3];
  const float* b_dec = (const float*)d_in[4];
  float* recon = (float*)d_out;
  float* codes = (float*)d_out + (size_t)B_ROWS * D_IN;

  // workspace layout
  char* ws = (char*)d_ws;
  float* nz_val = (float*)ws;                            ws += (size_t)B_ROWS * TOPK * 4;
  int*   nz_idx = (int*)ws;                              ws += (size_t)B_ROWS * TOPK * 4;
  unsigned short* xh = (unsigned short*)ws;              ws += (size_t)B_ROWS * D_IN * 2;
  unsigned short* xl = (unsigned short*)ws;              ws += (size_t)B_ROWS * D_IN * 2;
  unsigned short* wh = (unsigned short*)ws;              ws += (size_t)D_SAE * D_IN * 2;
  unsigned short* wl = (unsigned short*)ws;

  (void)hipFuncSetAttribute((const void*)sae_encode_wmma,
                            hipFuncAttributeMaxDynamicSharedMemorySize, 8 * REGION_BYTES);

  sae_cvt_x <<<(B_ROWS * D_IN / 4) / 256, 256, 0, stream>>>(x, xh, xl);
  sae_cvt_wt<<<dim3(D_IN / 32, D_SAE / 32), 256, 0, stream>>>(W_enc, wh, wl);
  sae_encode_wmma<<<dim3(D_SAE / 128, B_ROWS / 128), 256, 8 * REGION_BYTES, stream>>>(
      xh, xl, wh, wl, b_enc, codes);
  sae_topk  <<<B_ROWS, 256, 0, stream>>>(codes, nz_val, nz_idx);
  sae_decode<<<B_ROWS, 192, 0, stream>>>(W_dec, b_dec, nz_val, nz_idx, recon);
}
